// GMMLinear_13434657702654
// MI455X (gfx1250) — matrix operand order repaired and checked by reading, not compiled, not run on hardware
//
#include <hip/hip_runtime.h>
#include <math.h>

typedef __attribute__((ext_vector_type(16))) __bf16 bf16x16;
typedef __attribute__((ext_vector_type(8)))  float  v8f;
typedef __attribute__((ext_vector_type(4)))  unsigned int u32x4;
typedef __attribute__((ext_vector_type(8)))  int i32x8;
typedef __attribute__((ext_vector_type(4)))  int i32x4;

#define B_ROWS 8192
#define FIN    512
#define FOUT   512
#define MCOMP  8
#define NV     17   // 1 (xz*W) + 8 (nm*mean_p*W) + 8 (nm*exp(ls2)*W^2)

// dynamic-LDS double buffer layout (elements of __bf16)
#define BUF_ELEM 25600            // 51,200 bytes per buffer
#define XZ_E 0                    // 128 x 32
#define NM_E 4096                 // 128 x 32
#define B_E  8192                 // 17 x 32 x 32
#define BUF_BYTES 51200

__constant__ float kInvSqrt2Pi = 0.3989422804014327f;
__constant__ float kSqrtHalf   = 0.70710678118654752f;

union Frag16 { bf16x16 v; uint4 u[2]; };

// ---- TDM descriptor pack + issue (ISA cdna5 §8, 6-arg clang-23 builtin) ----
__device__ __forceinline__ void tdm_load(unsigned lds_byte_off, const void* gptr,
                                         unsigned tile_d0, unsigned tile_d1, unsigned tile_d2,
                                         unsigned long long stride0, unsigned long long stride1) {
    const unsigned long long ga = (unsigned long long)(uintptr_t)gptr;
    // group0: count=1 | lds_addr | global_addr[56:0] | type=2
    u32x4 g0 = { 1u,
                 lds_byte_off,
                 (unsigned)ga,
                 ((unsigned)((ga >> 32) & 0x1FFFFFFull)) | 0x80000000u };
    // group1: data_size=2B; tensor dims == tile dims (no OOB clip); 48-bit strides
    const unsigned td0 = tile_d0, td1 = tile_d1;
    i32x8 g1;
    g1[0] = (int)(1u << 16);                                            // data_size=1 -> 2 bytes
    g1[1] = (int)((td0 & 0xFFFFu) << 16);                               // tensor_dim0 [15:0]
    g1[2] = (int)(((td0 >> 16) & 0xFFFFu) | ((td1 & 0xFFFFu) << 16));   // td0 hi | td1 lo
    g1[3] = (int)(((td1 >> 16) & 0xFFFFu) | ((tile_d0 & 0xFFFFu) << 16)); // td1 hi | tile_dim0
    g1[4] = (int)((tile_d1 & 0xFFFFu) | ((tile_d2 & 0xFFFFu) << 16));   // tile_dim1 | tile_dim2
    g1[5] = (int)(unsigned)(stride0 & 0xFFFFFFFFull);                   // dim0_stride lo32
    g1[6] = (int)((unsigned)((stride0 >> 32) & 0xFFFFull) |
                  ((unsigned)(stride1 & 0xFFFFull) << 16));             // s0 hi16 | s1 lo16
    g1[7] = (int)(unsigned)((stride1 >> 16) & 0xFFFFFFFFull);           // s1 bits 47:16
    i32x4 g2 = { (int)tile_d2, 0, 0, 0 };                               // tensor_dim2
    i32x4 g3 = { 0, 0, 0, 0 };
    i32x8 g4 = { 0, 0, 0, 0, 0, 0, 0, 0 };
    __builtin_amdgcn_tensor_load_to_lds(g0, g1, g2, g3, g4, 0);
}

// ---------------- prep: softmax of log_weights (M=8) ----------------
__global__ void prep_softmax(const float* __restrict__ lw, float* __restrict__ smax) {
    if (threadIdx.x == 0) {
        float mx = -1e30f;
        for (int m = 0; m < MCOMP; ++m) mx = fmaxf(mx, lw[m]);
        float e[MCOMP]; float s = 0.f;
        for (int m = 0; m < MCOMP; ++m) { e[m] = __expf(lw[m] - mx); s += e[m]; }
        for (int m = 0; m < MCOMP; ++m) smax[m] = e[m] / s;
    }
}

// ---------------- prep: xz (nan->0), nm (nan indicator), empty flag --------
__global__ __launch_bounds__(128)
void prep_x(const float* __restrict__ x, __bf16* __restrict__ xz,
            __bf16* __restrict__ nm, unsigned* __restrict__ empty) {
    const int b = blockIdx.x;
    const int t = threadIdx.x;
    __shared__ int flag;
    if (t == 0) flag = 0;
    __syncthreads();
    int any = 0;
#pragma unroll
    for (int i = 0; i < 4; ++i) {
        const int f = t + i * 128;
        const float v = x[(size_t)b * FIN + f];
        const int isn = (v != v);
        any |= isn;
        xz[(size_t)b * FIN + f] = (__bf16)(isn ? 0.0f : v);
        nm[(size_t)b * FIN + f] = (__bf16)(isn ? 1.0f : 0.0f);
    }
    if (any) atomicOr(&flag, 1);
    __syncthreads();
    if (t == 0) empty[b] = (unsigned)flag;
}

// ---------------- prep: 17 variant weight matrices, bf16, [v][o][f] --------
__global__ __launch_bounds__(256)
void prep_w(const float* __restrict__ W, const float* __restrict__ mean_p,
            const float* __restrict__ log_std2, __bf16* __restrict__ bmat) {
    const size_t idx = (size_t)blockIdx.x * 256 + threadIdx.x;   // v*FOUT*FIN total
    const int f = (int)(idx % FIN);
    const size_t r = idx / FIN;
    const int o = (int)(r % FOUT);
    const int v = (int)(r / FOUT);
    const float w = W[(size_t)o * FIN + f];
    float val;
    if (v == 0)       val = w;
    else if (v <= 8)  val = mean_p[(size_t)(v - 1) * FIN + f] * w;
    else              val = __expf(log_std2[(size_t)(v - 9) * FIN + f]) * w * w;
    bmat[idx] = (__bf16)val;
}

// ---------------- main fused 17-GEMM + epilogue ----------------
// grid: (FOUT/32, B/128), block: 512 threads = 16 waves (wave32).
// wave tile: 16(b) x 16(o); waves arranged 8(b) x 2(o).
// All staging via Tensor Data Mover, double-buffered in dynamic LDS.
__global__ __launch_bounds__(512, 1)
void gmm_main(const __bf16* __restrict__ xz, const __bf16* __restrict__ nm,
              const __bf16* __restrict__ bmat, const float* __restrict__ bias,
              const float* __restrict__ smax, const unsigned* __restrict__ empty,
              float* __restrict__ out) {
    extern __shared__ __align__(16) __bf16 smem[];   // 2 * BUF_ELEM

    const int tid  = threadIdx.x;
    const int lane = tid & 31;
    const int wave = tid >> 5;      // 0..15
    const int wb   = wave >> 1;     // 0..7  b sub-tile
    const int wo   = wave & 1;      // 0..1  o sub-tile
    const int bBase = blockIdx.y * 128;
    const int oBase = blockIdx.x * 32;

    v8f acc[NV] = {};

    // TDM issue: 3 DMAs per k-chunk (xz tile, nm tile, 17-variant weight slab)
    auto issue = [&](int kc, int buf) {
        const unsigned base = (unsigned)(buf * BUF_BYTES);
        tdm_load(base + XZ_E * 2, xz + (size_t)bBase * FIN + kc,
                 32, 128, 0, FIN, 0);                              // 2D 128x32
        tdm_load(base + NM_E * 2, nm + (size_t)bBase * FIN + kc,
                 32, 128, 0, FIN, 0);                              // 2D 128x32
        tdm_load(base + B_E * 2, bmat + (size_t)oBase * FIN + kc,
                 32, 32, NV, FIN, (unsigned long long)FOUT * FIN); // 3D 17x32x32
    };

    if (wave == 0) issue(0, 0);
    int p = 0;

    for (int kc = 0; kc < FIN; kc += 32) {
        if (wave == 0) __builtin_amdgcn_s_wait_tensorcnt(0);
        __syncthreads();                       // chunk kc ready; buf p^1 compute done
        if (wave == 0 && kc + 32 < FIN) issue(kc + 32, p ^ 1);

        const __bf16* sXZ = smem + p * BUF_ELEM + XZ_E;
        const __bf16* sNM = smem + p * BUF_ELEM + NM_E;
        const __bf16* sB  = smem + p * BUF_ELEM + B_E;

        // A fragments (ISA 16-bit A layout: lanes 0-15 K=0..7,16..23; lanes 16-31 K=8..15,24..31)
        Frag16 aXZ, aNM;
        {
            const int row = wb * 16 + (lane & 15);
            const int k0  = (lane >> 4) * 8;
            aXZ.u[0] = *(const uint4*)(sXZ + row * 32 + k0);
            aXZ.u[1] = *(const uint4*)(sXZ + row * 32 + k0 + 16);
            aNM.u[0] = *(const uint4*)(sNM + row * 32 + k0);
            aNM.u[1] = *(const uint4*)(sNM + row * 32 + k0 + 16);
        }
        // B fragments (lanes 0-15: K=0..7,16..23 of col lane; lanes 16-31: K=8..15,24..31)
        const int colL = wo * 16 + (lane & 15);
        const int kb0  = (lane >> 4) * 8;
#pragma unroll
        for (int v = 0; v < NV; ++v) {
            Frag16 bf;
            bf.u[0] = *(const uint4*)(sB + v * 1024 + colL * 32 + kb0);
            bf.u[1] = *(const uint4*)(sB + v * 1024 + colL * 32 + kb0 + 16);
            const bf16x16 a = (v == 0) ? aXZ.v : aNM.v;
            acc[v] = __builtin_amdgcn_wmma_f32_16x16x32_bf16(
                false, a, false, bf.v, (short)0, acc[v], false, false);
        }
        p ^= 1;
    }

    // ---- epilogue: expected-ReLU mixture, fused on accumulator layout ----
    float sw[MCOMP];
#pragma unroll
    for (int m = 0; m < MCOMP; ++m) sw[m] = smax[m];

    const int col = oBase + wo * 16 + (lane & 15);
    const float bo = bias[col];
    const int rhalf = (lane >> 4) * 8;      // C layout: VGPR e -> rows e / e+8
#pragma unroll
    for (int e = 0; e < 8; ++e) {
        const int row = bBase + wb * 16 + rhalf + e;
        const float num0 = acc[0][e] + bo;
        float mix = 0.0f;
#pragma unroll
        for (int m = 0; m < MCOMP; ++m) {
            const float num = num0 + acc[1 + m][e];
            const float den = sqrtf(fmaxf(acc[9 + m][e], 0.0f));
            const float z   = num / (den + 1e-9f);
            const float pdf = kInvSqrt2Pi * __expf(-0.5f * z * z);
            const float nr  = den * (pdf + z * 0.5f * (1.0f + erff(z * kSqrtHalf)));
            mix = fmaf(sw[m], nr, mix);
        }
        out[(size_t)row * FOUT + col] = empty[row] ? mix : fmaxf(num0, 0.0f);
    }
}

extern "C" void kernel_launch(void* const* d_in, const int* in_sizes, int n_in,
                              void* d_out, int out_size, void* d_ws, size_t ws_size,
                              hipStream_t stream) {
    const float* x        = (const float*)d_in[0];
    const float* W        = (const float*)d_in[1];
    const float* bvec     = (const float*)d_in[2];
    const float* mean_p   = (const float*)d_in[3];
    const float* log_std2 = (const float*)d_in[4];
    const float* lw       = (const float*)d_in[5];
    float* out = (float*)d_out;

    char* ws = (char*)d_ws;
    __bf16*   xz    = (__bf16*)(ws);                        //  8,388,608 B
    __bf16*   nm    = (__bf16*)(ws + 8388608);              //  8,388,608 B
    __bf16*   bmat  = (__bf16*)(ws + 16777216);             //  8,912,896 B
    unsigned* empty = (unsigned*)(ws + 25690112);           //     32,768 B
    float*    smax  = (float*)(ws + 25722880);              //         32 B

    prep_softmax<<<1, 32, 0, stream>>>(lw, smax);
    prep_x<<<B_ROWS, 128, 0, stream>>>(x, xz, nm, empty);
    prep_w<<<(NV * FOUT * FIN) / 256, 256, 0, stream>>>(W, mean_p, log_std2, bmat);

    dim3 grid(FOUT / 32, B_ROWS / 128);
    gmm_main<<<grid, 512, 2 * BUF_BYTES, stream>>>(xz, nm, bmat, bvec, smax, empty, out);
}
